// ContMixT_20091857010822
// MI455X (gfx1250) — matrix operand order repaired and checked by reading, not compiled, not run on hardware
//
#include <hip/hip_runtime.h>
#include <math.h>

#define Bn   8
#define Cc   256
#define Hh   56
#define Ww   56
#define HW   (Hh*Ww)        // 3136
#define NPIX (Bn*HW)        // 25088
#define NPL  (Bn*Cc*HW)     // 6422528

typedef __attribute__((ext_vector_type(16))) __bf16 v16bf;
typedef __attribute__((ext_vector_type(8)))  float  v8f;

__device__ __forceinline__ unsigned short f2bf(float f) {
    unsigned u = __float_as_uint(f);
    u += 0x7FFFu + ((u >> 16) & 1u);          // round-to-nearest-even
    return (unsigned short)(u >> 16);
}

// ---------------------------------------------------------------------------
// Pre-pass: concat 3 frames -> bf16 [B][3C][H][W]
// ---------------------------------------------------------------------------
__global__ __launch_bounds__(256)
void cat_frames_bf16(const float* __restrict__ a, const float* __restrict__ b,
                     const float* __restrict__ c, unsigned short* __restrict__ out)
{
    int idx = blockIdx.x * 256 + threadIdx.x;          // < 8*768*HW (exact multiple)
    int bo = idx / (768 * HW);
    int r  = idx - bo * (768 * HW);
    int ci = r / HW;
    int hw = r - ci * HW;
    int plane = ci >> 8, cloc = ci & 255;
    const float* s = (plane == 0) ? a : ((plane == 1) ? b : c);
    out[idx] = f2bf(s[(bo * Cc + cloc) * HW + hw]);
}

// ---------------------------------------------------------------------------
// Pre-pass: weights [co][ci][3][3] f32 -> tap-major bf16 [tap][co][ci]
// ---------------------------------------------------------------------------
__global__ __launch_bounds__(256)
void reorder_wgt_bf16(const float* __restrict__ w, unsigned short* __restrict__ out,
                      int Cin)
{
    int idx = blockIdx.x * 256 + threadIdx.x;          // < 9*256*Cin (exact multiple)
    int tap = idx / (256 * Cin);
    int r   = idx - tap * (256 * Cin);
    int co  = r / Cin;
    int ci  = r - co * Cin;
    out[idx] = f2bf(w[(co * Cin + ci) * 9 + tap]);
}

// ---------------------------------------------------------------------------
// Implicit-GEMM dilated 3x3 conv, bf16 WMMA, fused bias+ReLU.
// inA: bf16 [B][Cin][HW]; wB: bf16 [9][256][Cin].
// Block: 256 threads = 8 wave32. Tile: 128 pixels x 64 out-channels.
// Wave w owns pixels [w*16,(w+1)*16); issues 4 WMMAs per K-chunk (16x16x32).
// B tile staged with GLOBAL_LOAD_ASYNC_TO_LDS_B32 (ASYNCcnt), overlapping the
// VALU im2col A staging; A next-chunk prefetched with global_prefetch_b8.
// ---------------------------------------------------------------------------
template<bool OUT_BF16>
__global__ __launch_bounds__(256)
void conv3x3_wmma(const unsigned short* __restrict__ inA,
                  const unsigned short* __restrict__ wB,
                  const float* __restrict__ bias,
                  float* __restrict__ outF, unsigned short* __restrict__ outB,
                  int Cin, int dil)
{
    __shared__ unsigned short shA[128 * 36];   // 128 px x 32 K (stride 36)
    __shared__ unsigned short shB[64 * 36];    // 64 n    x 32 K (stride 36)

    const int t    = threadIdx.x;
    const int lane = t & 31;
    const int wave = t >> 5;
    const int hl   = lane >> 4;
    const int l16  = lane & 15;

    const int pixbase = blockIdx.x * 128;
    const int cobase  = blockIdx.y * 64;

    // ---- fixed per-thread A staging slots (pixel decomposed ONCE) ----
    const int p   = t & 127;        // A pixel slot
    const int th  = t >> 7;         // A k parity: k = th, th+2, ..., th+30
    const int gp  = pixbase + p;
    const int pb  = gp / HW;
    const int phw = gp - pb * HW;
    const int ph  = phw / Ww;
    const int pw  = phw - ph * Ww;

    // ---- fixed per-thread B async slots: b32 word w16 of row n = nr0+16s ----
    const int w16 = t & 15;         // 32-bit word within a 64B row (2 bf16)
    const int nr0 = t >> 4;         // base row (0..15); rows n = nr0 + 16*s
    unsigned ldsB[4];
    #pragma unroll
    for (int s = 0; s < 4; ++s)
        ldsB[s] = (unsigned)(size_t)&shB[(nr0 + 16 * s) * 36 + 2 * w16];

    const int kchunks = Cin >> 5;
    v8f acc[4] = {{}, {}, {}, {}};

    for (int tap = 0; tap < 9; ++tap) {
        const int dy = (tap / 3 - 1) * dil;
        const int dx = (tap % 3 - 1) * dil;
        const int y  = ph + dy, x = pw + dx;
        const bool valid = ((unsigned)y < (unsigned)Hh) && ((unsigned)x < (unsigned)Ww);
        const int baseA = valid ? (pb * Cin + th) * HW + y * Ww + x : th * HW;

        // per-tap B global byte offsets (GVS: base SGPR64 + 32-bit VGPR offset)
        int gofB[4];
        #pragma unroll
        for (int s = 0; s < 4; ++s)
            gofB[s] = 2 * ((tap * 256 + cobase + nr0 + 16 * s) * Cin + 2 * w16);

        for (int kc = 0; kc < kchunks; ++kc) {
            __syncthreads();   // previous iteration's LDS reads complete

            // ---- async B tile: memory -> LDS, tracked by ASYNCcnt ----
            #pragma unroll
            for (int s = 0; s < 4; ++s) {
                asm volatile("global_load_async_to_lds_b32 %0, %1, %2"
                             :: "v"(ldsB[s]), "v"(gofB[s] + kc * 64), "s"(wB)
                             : "memory");
            }

            // ---- prefetch next A chunk (branchless clamp) ----
            const int kcn = (kc + 1 < kchunks) ? (kc + 1) : kc;
            __builtin_prefetch(&inA[baseA + kcn * 32 * HW], 0, 1);

            // ---- stage A: 16 u16 per thread, fixed LDS slots ----
            #pragma unroll
            for (int s = 0; s < 16; ++s) {
                unsigned short v = inA[baseA + (kc * 32 + 2 * s) * HW];
                shA[p * 36 + th + 2 * s] = valid ? v : (unsigned short)0;
            }

            asm volatile("s_wait_asynccnt 0x0" ::: "memory");
            __syncthreads();

            // ---- fragments: contiguous k = 16*hl + s in both A and B ----
            union { v16bf v; unsigned u[8]; } A, Bf;
            const int arow = wave * 16 + l16;
            #pragma unroll
            for (int v = 0; v < 8; ++v) {
                int ka = 2 * v + hl * 16;
                A.u[v] = *(const unsigned*)&shA[arow * 36 + ka];
            }
            #pragma unroll
            for (int nt = 0; nt < 4; ++nt) {
                #pragma unroll
                for (int v = 0; v < 8; ++v)
                    Bf.u[v] = *(const unsigned*)&shB[(nt * 16 + l16) * 36 + 2 * v + hl * 16];
                acc[nt] = __builtin_amdgcn_wmma_f32_16x16x32_bf16(
                    false, A.v, false, Bf.v, (short)0, acc[nt], false, false);
            }
        }
    }

    // ---- epilogue: bias + ReLU ----
    #pragma unroll
    for (int nt = 0; nt < 4; ++nt) {
        int co = cobase + nt * 16 + l16;
        float bi = bias[co];
        #pragma unroll
        for (int r = 0; r < 8; ++r) {
            int m   = r + hl * 8;
            int gpo = pixbase + wave * 16 + m;
            int b   = gpo / HW;
            int hw  = gpo - b * HW;
            float val = fmaxf(acc[nt][r] + bi, 0.f);
            if (OUT_BF16) outB[(b * Cc + co) * HW + hw] = f2bf(val);
            else          outF[(b * Cc + co) * HW + hw] = val;
        }
    }
}

// ---------------------------------------------------------------------------
__global__ __launch_bounds__(256)
void pool_mean(const float* __restrict__ gf2, const float* __restrict__ f_t,
               float* __restrict__ gpool, float* __restrict__ lpool)
{
    __shared__ float red[256];
    const float* src = (blockIdx.y == 0) ? gf2 : f_t;
    float* dst       = (blockIdx.y == 0) ? gpool : lpool;
    const float* base = src + (size_t)blockIdx.x * HW;
    float s = 0.f;
    for (int i = threadIdx.x; i < HW; i += 256) s += base[i];
    red[threadIdx.x] = s;
    __syncthreads();
    for (int off = 128; off > 0; off >>= 1) {
        if (threadIdx.x < off) red[threadIdx.x] += red[threadIdx.x + off];
        __syncthreads();
    }
    if (threadIdx.x == 0) dst[blockIdx.x] = red[0] * (1.f / HW);
}

// ---------------------------------------------------------------------------
__global__ __launch_bounds__(256)
void fc_kernel(const float* __restrict__ gpool, const float* __restrict__ lpool,
               const float* __restrict__ gw, const float* __restrict__ gb,
               const float* __restrict__ fc1w, const float* __restrict__ fc1b,
               const float* __restrict__ fc2w, const float* __restrict__ fc2b,
               float* __restrict__ wkout)
{
    __shared__ float shg[Bn * 256];
    __shared__ float shh[Bn * 512];
    const int t = threadIdx.x;

    for (int o = t; o < Bn * 256; o += 256) {        // 1x1 g_conv
        int b = o >> 8, co = o & 255;
        float s = gb[co];
        for (int ci = 0; ci < 256; ++ci) s += gpool[b * 256 + ci] * gw[co * 256 + ci];
        shg[o] = s;
    }
    __syncthreads();
    for (int o = t; o < Bn * 512; o += 256) {        // fc1 on [g_flat, local_pooled]
        int b = o >> 9, j = o & 511;
        float s = fc1b[j];
        const float* wr = fc1w + j * 512;
        for (int i = 0; i < 256; ++i) s += shg[b * 256 + i] * wr[i];
        for (int i = 0; i < 256; ++i) s += lpool[b * 256 + i] * wr[256 + i];
        shh[o] = s;
    }
    __syncthreads();
    for (int o = t; o < Bn * 2304; o += 256) {       // fc2 + SiLU
        int b = o / 2304, oo = o - b * 2304;
        float s = fc2b[oo];
        const float* wr = fc2w + oo * 512;
        const float* hr = shh + b * 512;
        for (int j = 0; j < 512; ++j) s += hr[j] * wr[j];
        wkout[o] = s / (1.f + __expf(-s));
    }
}

// ---------------------------------------------------------------------------
__global__ __launch_bounds__(256)
void dwconv_prev(const float* __restrict__ f_t, const float* __restrict__ f_tm1,
                 const float* __restrict__ f_tm2, const float* __restrict__ wk,
                 float* __restrict__ mod, float* __restrict__ prev)
{
    const int bc = blockIdx.x;              // b*256 + c
    float k[9];
    #pragma unroll
    for (int i = 0; i < 9; ++i) k[i] = wk[bc * 9 + i];
    const float* src = f_t + (size_t)bc * HW;
    for (int p = threadIdx.x; p < HW; p += 256) {
        int h = p / Ww, w = p - h * Ww;
        float s = 0.f;
        #pragma unroll
        for (int ky = 0; ky < 3; ++ky) {
            int y = h + ky - 1;
            if ((unsigned)y >= (unsigned)Hh) continue;
            #pragma unroll
            for (int kx = 0; kx < 3; ++kx) {
                int x = w + kx - 1;
                if ((unsigned)x >= (unsigned)Ww) continue;
                s += k[ky * 3 + kx] * src[y * Ww + x];
            }
        }
        size_t idx = (size_t)bc * HW + p;
        mod[idx]  = s;
        prev[idx] = 0.5f * (f_tm1[idx] + f_tm2[idx]);
    }
}

// ---------------------------------------------------------------------------
__global__ __launch_bounds__(256)
void alpha_kernel(const float* __restrict__ mod, const float* __restrict__ prev,
                  const float* __restrict__ aw, const float* __restrict__ ab,
                  float* __restrict__ alpha)
{
    int gp = blockIdx.x * 256 + threadIdx.x;
    if (gp >= NPIX) return;
    int b = gp / HW, hw = gp - b * HW;
    float s = ab[0];
    for (int c = 0; c < 256; ++c) {
        size_t idx = ((size_t)(b * Cc + c)) * HW + hw;
        s += aw[c] * mod[idx] + aw[256 + c] * prev[idx];
    }
    alpha[gp] = 0.3f + 0.4f / (1.f + __expf(-s));
}

// ---------------------------------------------------------------------------
__global__ __launch_bounds__(256)
void fuse_kernel(const float* __restrict__ mod, const float* __restrict__ prev,
                 const float* __restrict__ alpha, float* __restrict__ out)
{
    int i = blockIdx.x * 256 + threadIdx.x;
    if (i >= NPL) return;
    int bc = i / HW, hw = i - bc * HW;
    int b = bc >> 8;
    float a = alpha[b * HW + hw];
    out[i] = a * mod[i] + (1.f - a) * prev[i];
}

// ---------------------------------------------------------------------------
extern "C" void kernel_launch(void* const* d_in, const int* in_sizes, int n_in,
                              void* d_out, int out_size, void* d_ws, size_t ws_size,
                              hipStream_t stream)
{
    const float* f_tm2 = (const float*)d_in[0];
    const float* f_tm1 = (const float*)d_in[1];
    const float* f_t   = (const float*)d_in[2];
    const float* w1    = (const float*)d_in[3];
    const float* b1    = (const float*)d_in[4];
    const float* w2    = (const float*)d_in[5];
    const float* b2    = (const float*)d_in[6];
    const float* gw    = (const float*)d_in[7];
    const float* gb    = (const float*)d_in[8];
    const float* fc1w  = (const float*)d_in[9];
    const float* fc1b  = (const float*)d_in[10];
    const float* fc2w  = (const float*)d_in[11];
    const float* fc2b  = (const float*)d_in[12];
    const float* aw    = (const float*)d_in[13];
    const float* ab    = (const float*)d_in[14];

    // ---- workspace carve-up (with reuse) ----
    char* ws = (char*)d_ws;
    unsigned short* catb = (unsigned short*)ws;          // 8*768*HW bf16 = 38,535,168 B
    float*          modb = (float*)ws;                   // reused after conv1 (25.7 MB fits)
    char* w1p = ws + (size_t)Bn * 768 * HW * 2;
    unsigned short* gf1b = (unsigned short*)w1p;         // conv1 out bf16: 12,845,056 B
    char* w2p = w1p + (size_t)NPL * 2;
    float*          gf2  = (float*)w2p;                  // conv2 out f32 (reused as prev)
    char* w3p = w2p + (size_t)NPL * 4;
    unsigned short* wb1  = (unsigned short*)w3p;         // 9*256*768 bf16
    char* w4p = w3p + (size_t)9 * 256 * 768 * 2;
    unsigned short* wb2  = (unsigned short*)w4p;         // 9*256*256 bf16
    char* w5p = w4p + (size_t)9 * 256 * 256 * 2;
    float* gpool    = (float*)w5p;
    float* lpool    = gpool + Bn * Cc;
    float* wkbuf    = lpool + Bn * Cc;                   // 8*2304
    float* alphabuf = wkbuf + Bn * Cc * 9;               // 25088

    // ---- pre-passes: bf16 concat + tap-major bf16 weights ----
    cat_frames_bf16<<<(Bn * 768 * HW) / 256, 256, 0, stream>>>(f_tm2, f_tm1, f_t, catb);
    reorder_wgt_bf16<<<(9 * 256 * 768) / 256, 256, 0, stream>>>(w1, wb1, 768);
    reorder_wgt_bf16<<<(9 * 256 * 256) / 256, 256, 0, stream>>>(w2, wb2, 256);

    // ---- WMMA convs ----
    conv3x3_wmma<true><<<dim3(NPIX / 128, Cc / 64), 256, 0, stream>>>(
        catb, wb1, b1, nullptr, gf1b, 3 * Cc, 2);
    conv3x3_wmma<false><<<dim3(NPIX / 128, Cc / 64), 256, 0, stream>>>(
        gf1b, wb2, b2, gf2, nullptr, Cc, 4);

    // ---- tail ----
    pool_mean<<<dim3(Bn * Cc, 2), 256, 0, stream>>>(gf2, f_t, gpool, lpool);
    fc_kernel<<<1, 256, 0, stream>>>(gpool, lpool, gw, gb, fc1w, fc1b, fc2w, fc2b, wkbuf);
    dwconv_prev<<<Bn * Cc, 256, 0, stream>>>(f_t, f_tm1, f_tm2, wkbuf, modb, gf2);
    alpha_kernel<<<(NPIX + 255) / 256, 256, 0, stream>>>(modb, gf2, aw, ab, alphabuf);
    fuse_kernel<<<(NPL + 255) / 256, 256, 0, stream>>>(modb, gf2, alphabuf, (float*)d_out);
}
